// Kernelizer_43319040147946
// MI455X (gfx1250) — matrix operand order repaired and checked by reading, not compiled, not run on hardware
//
#include <hip/hip_runtime.h>
#include <cstdint>

typedef _Float16 v16h __attribute__((ext_vector_type(16)));
typedef _Float16 v8h  __attribute__((ext_vector_type(8)));
typedef float    v8f  __attribute__((ext_vector_type(8)));

#define CH     1024
#define NST    16
#define LSEQ   4096
#define TCH    64      // chunk length
#define NCHK   64      // LSEQ / TCH

static __device__ __forceinline__ v8f wmma_f16(v16h a, v16h b, v8f c) {
  // D = A(16x32 f16) * B(32x16 f16) + C(16x16 f32)
  return __builtin_amdgcn_wmma_f32_16x16x32_f16(
      /*neg_a=*/false, a, /*neg_b=*/false, b,
      /*c_mod=*/(short)0, c, /*reuse_a=*/false, /*reuse_b=*/false);
}

// A-matrix fragment (16x32 f16) from global, row-major with rowStride halves.
// Layout used here: lane row m = lane%16; half h = lane/16;
//   k = kt*32 + h*16 + [0..15] contiguous (matches B fragments)
static __device__ __forceinline__ v16h loadA_g(const _Float16* base, int m, int kt,
                                               int rowStride, int half) {
  const _Float16* p0 = base + m * rowStride + kt * 32 + half * 16;
  union { v16h v; v8h h[2]; } r;
  r.h[0] = *(const v8h*)p0;        // k = kt*32 + h*16 + 0..7
  r.h[1] = *(const v8h*)(p0 + 8);  // k = kt*32 + h*16 + 8..15
  return r.v;
}

// B-matrix fragment (32x16 f16) from f32 LDS sequence u_sh[chunk*64 + k].
// Layout used here: lane col n = lane%16; k = (lane/16)*16 + [0..15] contiguous.
static __device__ __forceinline__ v16h loadB_u(const float* u_sh, int n, int kt, int half) {
  const float* p = u_sh + n * 64 + kt * 32 + half * 16;
  v16h r;
#pragma unroll
  for (int i = 0; i < 16; ++i) r[i] = (_Float16)p[i];
  return r;
}

// B-matrix fragment (32x16 f16) from LDS halves S_sh[chunk*32 + k].
static __device__ __forceinline__ v16h loadB_s(const _Float16* S_sh, int n, int half) {
  const _Float16* p = S_sh + n * 32 + half * 16;
  union { v16h v; v8h h[2]; } r;
  r.h[0] = *(const v8h*)p;
  r.h[1] = *(const v8h*)(p + 8);
  return r.v;
}

// ---------------------------------------------------------------------------
// Kernel 1: per-channel precompute of WMMA operand matrices (f16) + z^T (f32).
//   WinPk [c][row 0..31][t 0..63] : rows 0-15 Re(z^(64-t)), rows 16-31 Im(z^(64-t))
//   WoutPk[c][t 0..63][k 0..31]   : k 0-15 Re(Eh*z^t), k 16-31 -Im(Eh*z^t)
//   Ktri  [c][t][t']              : k[t-t'] lower-triangular, k[m]=Re(sum_n Eh*z^m)
//   zT    [c][0..31]              : Re(z^64) (0-15), Im(z^64) (16-31)
// ---------------------------------------------------------------------------
__global__ void __launch_bounds__(64)
precompute_kernel(const float* __restrict__ A, const float* __restrict__ log_dt,
                  const float* __restrict__ E,
                  _Float16* __restrict__ WinPk, _Float16* __restrict__ WoutPk,
                  _Float16* __restrict__ Ktri, float* __restrict__ zT) {
  const int c = blockIdx.x;
  const int t = threadIdx.x;   // 0..63
  __shared__ float s_dar[NST], s_dai[NST], s_Eh[NST];
  __shared__ float s_k[TCH];

  if (t < NST) {
    const int n = t;
    const float a0 = A[(c * NST + n) * 2 + 0];
    const float ai = A[(c * NST + n) * 2 + 1];
    const float ar = -log1pf(expf(a0));       // -softplus
    const float dt = expf(log_dt[c * NST + n]);
    s_dar[n] = dt * ar;
    s_dai[n] = dt * ai;
    s_Eh[n]  = E[c * NST + n] * dt;
  }
  __syncthreads();

  float ksum = 0.f;
#pragma unroll 4
  for (int n = 0; n < NST; ++n) {
    const float dar = s_dar[n], dai = s_dai[n], Eh = s_Eh[n];
    // Win: z^(64 - t)
    const float m1 = (float)(TCH - t);
    const float e1 = expf(dar * m1);
    WinPk[((size_t)c * 32 + n) * TCH + t]      = (_Float16)(e1 * cosf(dai * m1));
    WinPk[((size_t)c * 32 + 16 + n) * TCH + t] = (_Float16)(e1 * sinf(dai * m1));
    // Wout: Eh * z^t  (real part, minus imag part packed in k 16..31)
    const float e0 = expf(dar * (float)t);
    const float cr = e0 * cosf(dai * (float)t);
    const float ci = e0 * sinf(dai * (float)t);
    WoutPk[((size_t)c * TCH + t) * 32 + n]      = (_Float16)(Eh * cr);
    WoutPk[((size_t)c * TCH + t) * 32 + 16 + n] = (_Float16)(-Eh * ci);
    ksum += Eh * cr;                            // k[t]
  }
  s_k[t] = ksum;
  if (t < NST) {
    const int n = t;
    const float e = expf(s_dar[n] * (float)TCH);
    zT[(size_t)c * 32 + n]      = e * cosf(s_dai[n] * (float)TCH);
    zT[(size_t)c * 32 + 16 + n] = e * sinf(s_dai[n] * (float)TCH);
  }
  __syncthreads();

  _Float16* row = Ktri + ((size_t)c * TCH + t) * TCH;
#pragma unroll 8
  for (int tp = 0; tp < TCH; ++tp)
    row[tp] = (tp <= t) ? (_Float16)s_k[t - tp] : (_Float16)0.f;
}

// ---------------------------------------------------------------------------
// Kernel 2: one block per (b,c). Async-stream u into LDS, then:
//   Phase A : P = Win @ U          (WMMA, all chunks in parallel)
//   Phase B : 64-step complex scan (wave 0)
//   Phase C+D: Y = Ktri @ U + WoutPk @ [S_re;S_im]   (chained WMMA)
// ---------------------------------------------------------------------------
__global__ void __launch_bounds__(128)
ssm_conv_kernel(const float* __restrict__ u, const _Float16* __restrict__ WinPk,
                const _Float16* __restrict__ WoutPk, const _Float16* __restrict__ Ktri,
                const float* __restrict__ zT, float* __restrict__ y) {
  __shared__ __align__(16) float    u_sh[LSEQ];        // 16 KB input slice (f32)
  __shared__ __align__(16) float    P_sh[NCHK * 32];   //  8 KB P[j][n(re)/16+n(im)]
  __shared__ __align__(16) _Float16 S_sh[NCHK * 32];   //  4 KB S[j][n(re)/16+n(im)]

  const int bc   = blockIdx.x;           // b*CH + c
  const int c    = bc & (CH - 1);
  const int tid  = threadIdx.x;
  const int lane = tid & 31;
  const int wave = tid >> 5;             // 0..3, owns column-tile jt = wave
  const int jt   = wave;
  const int mrow = lane & 15;
  const int half = lane >> 4;

  const _Float16* WinC  = WinPk  + (size_t)c * 32 * TCH;
  const _Float16* WoutC = WoutPk + (size_t)c * TCH * 32;
  const _Float16* KtriC = Ktri   + (size_t)c * TCH * TCH;
  __builtin_prefetch(KtriC, 0, 3);       // global_prefetch_b8 into L2/L0
  __builtin_prefetch(WinC, 0, 3);

  // ---- async stream: global -> LDS (16 B per lane per issue, 8 issues) ----
  const uint64_t ubase = (uint64_t)(uintptr_t)(u + (size_t)bc * LSEQ);
#pragma unroll
  for (int i = 0; i < 8; ++i) {
    const int chunk16 = i * 128 + tid;   // 16-byte chunk id, 0..1023
    const uint32_t ldsoff = (uint32_t)(uintptr_t)(&u_sh[chunk16 * 4]);
    const uint64_t gaddr  = ubase + (uint64_t)chunk16 * 16;
    asm volatile("global_load_async_to_lds_b128 %0, %1, off"
                 :: "v"(ldsoff), "v"(gaddr) : "memory");
  }
  asm volatile("s_wait_asynccnt 0" ::: "memory");
  __syncthreads();

  // ---- shared B fragments of U for this wave's 16 chunks ----
  const v16h bu0 = loadB_u(u_sh, jt * 16 + mrow, 0, half);
  const v16h bu1 = loadB_u(u_sh, jt * 16 + mrow, 1, half);

  // ---- Phase A: P_re/P_im = Win_re/Win_im @ U  (K = 64 -> two k-tiles) ----
  {
    v8f pr = {}, pi = {};
    v16h a;
    a = loadA_g(WinC, mrow,      0, TCH, half); pr = wmma_f16(a, bu0, pr);
    a = loadA_g(WinC, mrow,      1, TCH, half); pr = wmma_f16(a, bu1, pr);
    a = loadA_g(WinC, 16 + mrow, 0, TCH, half); pi = wmma_f16(a, bu0, pi);
    a = loadA_g(WinC, 16 + mrow, 1, TCH, half); pi = wmma_f16(a, bu1, pi);
    const int j = jt * 16 + mrow;        // C/D layout: N = lane%16
#pragma unroll
    for (int r = 0; r < 8; ++r) {        // M = r + 8*half
      const int M = r + 8 * half;
      P_sh[j * 32 + M]      = pr[r];
      P_sh[j * 32 + 16 + M] = pi[r];
    }
  }
  __syncthreads();

  // ---- Phase B: sequential complex diagonal scan over 64 chunks (wave 0) ----
  if (wave == 0 && lane < NST) {
    const int n = lane;
    const float zr = zT[(size_t)c * 32 + n];
    const float zi = zT[(size_t)c * 32 + 16 + n];
    float sr = 0.f, si = 0.f;
    for (int j = 0; j < NCHK; ++j) {
      S_sh[j * 32 + n]      = (_Float16)sr;   // state *entering* chunk j
      S_sh[j * 32 + 16 + n] = (_Float16)si;
      const float pr = P_sh[j * 32 + n];
      const float pi = P_sh[j * 32 + 16 + n];
      const float nsr = zr * sr - zi * si + pr;
      const float nsi = zr * si + zi * sr + pi;
      sr = nsr; si = nsi;
    }
  }
  __syncthreads();

  // ---- Phase C+D: Y = Ktri@U + WoutPk@[S_re;S_im], chained into one acc ----
  const v16h bs = loadB_s(S_sh, jt * 16 + mrow, half);
  float* ybase = y + (size_t)bc * LSEQ;
#pragma unroll
  for (int tt = 0; tt < 4; ++tt) {       // row-tiles of t
    v8f acc = {};
    v16h a;
    a = loadA_g(KtriC, tt * 16 + mrow, 0, TCH, half); acc = wmma_f16(a, bu0, acc);
    a = loadA_g(KtriC, tt * 16 + mrow, 1, TCH, half); acc = wmma_f16(a, bu1, acc);
    a = loadA_g(WoutC, tt * 16 + mrow, 0, 32,  half); acc = wmma_f16(a, bs,  acc);
    // store: l = (jt*16 + mrow)*64 + tt*16 + 8*half + r  -> two float4s
    const int l0 = (jt * 16 + mrow) * 64 + tt * 16 + 8 * half;
    *(float4*)(ybase + l0)     = make_float4(acc[0], acc[1], acc[2], acc[3]);
    *(float4*)(ybase + l0 + 4) = make_float4(acc[4], acc[5], acc[6], acc[7]);
  }
}

// ---------------------------------------------------------------------------
extern "C" void kernel_launch(void* const* d_in, const int* in_sizes, int n_in,
                              void* d_out, int out_size, void* d_ws, size_t ws_size,
                              hipStream_t stream) {
  const float* input  = (const float*)d_in[0];   // (B, 1024, 4096)
  const float* A      = (const float*)d_in[1];   // (1024, 16, 2)
  const float* log_dt = (const float*)d_in[4];   // (1024, 16)
  const float* E      = (const float*)d_in[5];   // (1024, 16)
  float* out = (float*)d_out;

  const int Bsz = in_sizes[0] / (CH * LSEQ);     // 16

  // workspace carve-up (16 MB + 128 KB; f16 operand matrices stay L2-resident)
  _Float16* WinPk  = (_Float16*)d_ws;                      // CH*32*64
  _Float16* WoutPk = WinPk  + (size_t)CH * 32 * TCH;       // CH*64*32
  _Float16* Ktri   = WoutPk + (size_t)CH * TCH * 32;       // CH*64*64
  float*    zT     = (float*)(Ktri + (size_t)CH * TCH * TCH); // CH*32

  precompute_kernel<<<CH, 64, 0, stream>>>(A, log_dt, E, WinPk, WoutPk, Ktri, zT);
  ssm_conv_kernel<<<Bsz * CH, 128, 0, stream>>>(input, WinPk, WoutPk, Ktri, zT, out);
}
